// MyLSTM_46591805227535
// MI455X (gfx1250) — matrix operand order repaired and checked by reading, not compiled, not run on hardware
//
#include <hip/hip_runtime.h>
#include <hip/hip_bf16.h>

// Problem dims (fixed by the reference)
#define B_   16
#define S_   1024
#define V_   8000
#define H_   1024
#define G4H  4096
#define NBLK 64          // persistent blocks for the recurrent kernel

typedef unsigned short u16;
typedef __attribute__((ext_vector_type(16))) __bf16 bf16x16;
typedef __attribute__((ext_vector_type(8)))  float  v8f;

union FragU { uint4 u[2]; bf16x16 v; };

__device__ __forceinline__ u16 f2bf(float f) {
    unsigned u = __float_as_uint(f);
    u += 0x7FFFu + ((u >> 16) & 1u);          // round-to-nearest-even
    return (u16)(u >> 16);
}

// Scheduling fence: forbid the machine scheduler from sinking loads across it
// (prevents re-serializing the batched A-fragment clause into load->wait->wmma).
__device__ __forceinline__ void sched_fence() {
#if __has_builtin(__builtin_amdgcn_sched_barrier)
    __builtin_amdgcn_sched_barrier(0);
#endif
}

// A-fragment (16x32 bf16, row-major source): lane holds row m=L%16,
// K chunks {k0+hi*8..+7} and {k0+16+hi*8..+7}  (hi = L>=16)
__device__ __forceinline__ bf16x16 load_frag_A(const u16* __restrict__ row, int hi, int k0) {
    const u16* r = row + k0 + hi * 8;
    FragU f;
    f.u[0] = *(const uint4*)(r);
    f.u[1] = *(const uint4*)(r + 16);
    return f.v;
}

// B-fragment (32x16 bf16) from TRANSPOSED weights [N][K] (or an LDS tile):
// lane holds column n=L%16 (folded into rowT), K chunks {k0+hi*8..+7} and {k0+16+hi*8..+7}.
__device__ __forceinline__ bf16x16 load_frag_B(const u16* __restrict__ rowT, int hi, int k0) {
    const u16* r = rowT + k0 + hi * 8;
    FragU f;
    f.u[0] = *(const uint4*)(r);
    f.u[1] = *(const uint4*)(r + 16);
    return f.v;
}

__device__ __forceinline__ v8f wmma_bf16(bf16x16 a, bf16x16 b, v8f c) {
    return __builtin_amdgcn_wmma_f32_16x16x32_bf16(false, a, false, b, (short)0, c, false, false);
}

__device__ __forceinline__ v8f vzero8() {
    v8f z = {0.f, 0.f, 0.f, 0.f, 0.f, 0.f, 0.f, 0.f};
    return z;
}

__device__ __forceinline__ float fsigmoid(float x) { return 1.f / (1.f + __expf(-x)); }

// CDNA5 async global->LDS copy (ASYNCcnt-tracked). LDS destination offset is
// the low 32 bits of the generic pointer.
__device__ __forceinline__ void async_ld_b128(void* lds_dst, const void* gsrc) {
    unsigned loff = (unsigned)(uintptr_t)lds_dst;
    asm volatile("global_load_async_to_lds_b128 %0, %1, off"
                 :: "v"(loff), "v"(gsrc)
                 : "memory");
}
__device__ __forceinline__ void wait_async0() {
    asm volatile("s_wait_asynccnt 0x0" ::: "memory");
}

// ---------------------------------------------------------------------------
// Prep kernels
// ---------------------------------------------------------------------------
__global__ void transpose_bf16_kernel(const float* __restrict__ src, u16* __restrict__ dst,
                                      int K, int N) {
    // src[K][N] fp32 -> dst[N][K] bf16
    size_t i = (size_t)blockIdx.x * blockDim.x + threadIdx.x;
    size_t total = (size_t)K * N;
    if (i >= total) return;
    size_t n = i / (size_t)K;
    size_t k = i - n * (size_t)K;
    dst[i] = f2bf(src[k * (size_t)N + n]);
}

__global__ void bsum_kernel(const float* __restrict__ a, const float* __restrict__ b,
                            float* __restrict__ o) {
    int i = blockIdx.x * blockDim.x + threadIdx.x;
    if (i < G4H) o[i] = a[i] + b[i];
}

// ---------------------------------------------------------------------------
// Recurrent LSTM: persistent kernel, grid-wide barrier per timestep.
// Block j owns h-columns [16j, 16j+16). 16 waves: wave = kq*4 + g computes the
// 16x16 tile of gate g over K quarter kq (8 WMMAs). B-fragments of W_hh are
// register-resident for the whole sequence; all 8 A-fragments are fetched in
// one pinned clause (single L2 round-trip per step) before the WMMA chain.
// ---------------------------------------------------------------------------
__global__ void __launch_bounds__(512)
lstm_seq_kernel(const int*   __restrict__ ids,     // [B][S]
                const float* __restrict__ W_ih,    // [V][4H] fp32
                const u16*   __restrict__ whh_t,   // [4H][H] bf16 (transposed)
                const float* __restrict__ bsum,    // [4H]
                u16*         __restrict__ hbuf,    // [2][B][H] bf16, zero-init
                u16*         __restrict__ hs,      // [B][S][H] bf16
                int*         __restrict__ bar)     // zero-init
{
    __shared__ float parts[16 * 256];              // 16 waves x (16b x 16col)

    const int j    = blockIdx.x;
    const int tid  = threadIdx.x;
    const int wave = tid >> 5;
    const int lane = tid & 31;
    const int g    = wave & 3;                     // gate tile (i,f,g,o)
    const int kq   = wave >> 2;                    // K quarter
    const int m    = lane & 15;
    const int hi   = lane >> 4;
    const int kbase = kq * 256;

    // Loop-invariant B fragments: transposed W_hh row = gate-col (g*H + 16j + m)
    const u16* brow = whh_t + (size_t)(g * H_ + j * 16 + m) * H_;
    bf16x16 bfrag[8];
#pragma unroll
    for (int kk = 0; kk < 8; ++kk) bfrag[kk] = load_frag_B(brow, hi, kbase + kk * 32);

    // elementwise identity (tid < 256): b = tid>>4, col-in-block = tid&15
    const int eb = tid >> 4;
    const int ec = tid & 15;
    float bs0 = 0.f, bs1 = 0.f, bs2 = 0.f, bs3 = 0.f, creg = 0.f;
    if (tid < 256) {
        int col = j * 16 + ec;
        bs0 = bsum[col];
        bs1 = bsum[H_ + col];
        bs2 = bsum[2 * H_ + col];
        bs3 = bsum[3 * H_ + col];
    }

    for (int t = 0; t < S_; ++t) {
        // ---- prefetch x_t = W_ih[id] rows (independent of h, overlaps WMMA)
        float x0 = 0.f, x1 = 0.f, x2 = 0.f, x3 = 0.f;
        if (tid < 256) {
            int id = ids[eb * S_ + t];
            const float* wr = W_ih + (size_t)id * G4H + j * 16 + ec;
            x0 = wr[0];
            x1 = wr[H_];
            x2 = wr[2 * H_];
            x3 = wr[3 * H_];
        }
        sched_fence();                              // keep x loads issued early

        // ---- batch-load all 8 A-fragments (pinned: one 16-load clause)
        const u16* arow = hbuf + (size_t)(t & 1) * (B_ * H_) + (size_t)m * H_;
        bf16x16 af[8];
#pragma unroll
        for (int kk = 0; kk < 8; ++kk) af[kk] = load_frag_A(arow, hi, kbase + kk * 32);
        sched_fence();                              // all loads before any WMMA

        // ---- h @ W_hh partial tile (this wave's K quarter)
        v8f acc = vzero8();
#pragma unroll
        for (int kk = 0; kk < 8; ++kk) acc = wmma_bf16(af[kk], bfrag[kk], acc);
        sched_fence();

        // ---- reduce partials through LDS
#pragma unroll
        for (int r = 0; r < 8; ++r)
            parts[wave * 256 + (r + hi * 8) * 16 + m] = acc[r];
        __syncthreads();

        if (tid < 256) {
            float pi = parts[0 * 256 + tid] + parts[4 * 256 + tid] +
                       parts[8 * 256 + tid] + parts[12 * 256 + tid] + x0 + bs0;
            float pf = parts[1 * 256 + tid] + parts[5 * 256 + tid] +
                       parts[9 * 256 + tid] + parts[13 * 256 + tid] + x1 + bs1;
            float pg = parts[2 * 256 + tid] + parts[6 * 256 + tid] +
                       parts[10 * 256 + tid] + parts[14 * 256 + tid] + x2 + bs2;
            float po = parts[3 * 256 + tid] + parts[7 * 256 + tid] +
                       parts[11 * 256 + tid] + parts[15 * 256 + tid] + x3 + bs3;
            float iv = fsigmoid(pi);
            float fv = fsigmoid(pf);
            float gv = tanhf(pg);
            float ov = fsigmoid(po);
            creg = fv * creg + iv * gv;
            float h = ov * tanhf(creg);
            u16 hb = f2bf(h);
            int col = j * 16 + ec;
            hbuf[(size_t)((t + 1) & 1) * (B_ * H_) + (size_t)eb * H_ + col] = hb;
            hs[((size_t)eb * S_ + t) * H_ + col] = hb;
        }

        // ---- device-wide barrier (monotonic counter, target = (t+1)*NBLK)
        __syncthreads();
        if (tid == 0) {
            __threadfence();                        // release h writes to L2
            __hip_atomic_fetch_add(bar, 1, __ATOMIC_RELEASE, __HIP_MEMORY_SCOPE_AGENT);
            int target = (t + 1) * NBLK;
            while (__hip_atomic_load(bar, __ATOMIC_ACQUIRE, __HIP_MEMORY_SCOPE_AGENT) < target)
                __builtin_amdgcn_s_sleep(1);
        }
        __syncthreads();
        __threadfence();                            // acquire: drop stale L0 lines
    }
}

// ---------------------------------------------------------------------------
// Output projection: [16384,1024] x [1024,8000] + b_out -> fp32 logits.
// Block tile M=128 (8 waves) x N=80. W_out B-tiles are staged through LDS with
// CDNA5 async global->LDS copies, double-buffered over 8 K-chunks of 128, so
// all 8 waves share one copy of B (8x less L2 traffic). A-fragments batched.
// ---------------------------------------------------------------------------
__global__ void __launch_bounds__(256)
out_gemm_kernel(const u16*  __restrict__ hs,       // [16384][1024] bf16
                const u16*  __restrict__ wout_t,   // [8000][1024] bf16 (transposed)
                const float* __restrict__ b_out,   // [8000]
                float*       __restrict__ out)     // [16384][8000]
{
    __shared__ u16 bstage[2][80 * 128];            // 2 x 20 KB K-chunk of B

    const int tid  = threadIdx.x;
    const int wave = tid >> 5;
    const int lane = tid & 31;
    const int m    = lane & 15;
    const int hi   = lane >> 4;
    const int row0 = blockIdx.y * 128 + wave * 16;
    const int n0   = blockIdx.x * 80;

    const u16* arow = hs + (size_t)(row0 + m) * H_;
    float bias[5];
#pragma unroll
    for (int nt = 0; nt < 5; ++nt) bias[nt] = b_out[n0 + nt * 16 + m];

    v8f acc[5];
#pragma unroll
    for (int nt = 0; nt < 5; ++nt) acc[nt] = vzero8();

    // stage K-chunk kc into buffer (kc&1): 80 rows x 128 cols bf16 = 1280 x b128
    auto issue_stage = [&](int kc) {
#pragma unroll
        for (int i = 0; i < 5; ++i) {
            int e   = tid + i * 256;               // 0..1279
            int row = e >> 4;                      // 0..79
            int c8  = (e & 15) << 3;               // col chunk * 8
            const u16* gsrc = wout_t + (size_t)(n0 + row) * H_ + kc * 128 + c8;
            async_ld_b128(&bstage[kc & 1][row * 128 + c8], gsrc);
        }
    };

    issue_stage(0);
    wait_async0();
    __syncthreads();

    for (int kc = 0; kc < 8; ++kc) {
        if (kc + 1 < 8) issue_stage(kc + 1);       // overlaps with compute below

        const u16* bbase = &bstage[kc & 1][0];
        bf16x16 a4[4];
#pragma unroll
        for (int kk = 0; kk < 4; ++kk) a4[kk] = load_frag_A(arow, hi, kc * 128 + kk * 32);
        sched_fence();                             // batch A loads before WMMAs
#pragma unroll
        for (int kk = 0; kk < 4; ++kk) {
#pragma unroll
            for (int nt = 0; nt < 5; ++nt) {
                bf16x16 b = load_frag_B(bbase + (nt * 16 + m) * 128, hi, kk * 32);
                acc[nt] = wmma_bf16(a4[kk], b, acc[nt]);
            }
        }

        __syncthreads();                           // everyone done reading buf[kc&1]
        if (kc + 1 < 8) {
            wait_async0();                         // chunk kc+1 landed in LDS
            __syncthreads();
        }
    }

#pragma unroll
    for (int nt = 0; nt < 5; ++nt) {
        int n = n0 + nt * 16 + m;
#pragma unroll
        for (int r = 0; r < 8; ++r)
            out[(size_t)(row0 + r + hi * 8) * V_ + n] = acc[nt][r] + bias[nt];
    }
}

// ---------------------------------------------------------------------------
// Workspace layout (bytes)
// ---------------------------------------------------------------------------
#define OFF_WHHT  ((size_t)0)                          // 4096*1024*2 = 8388608
#define OFF_WOUTT ((size_t)8388608)                    // 8000*1024*2 = 16384000
#define OFF_BSUM  ((size_t)(8388608 + 16384000))       // 4096*4 = 16384
#define OFF_HBUF  (OFF_BSUM + 16384)                   // 2*16*1024*2 = 65536
#define OFF_BAR   (OFF_HBUF + 65536)                   // 256
#define OFF_HS    (OFF_BAR + 256)                      // 16*1024*1024*2 = 33554432

extern "C" void kernel_launch(void* const* d_in, const int* in_sizes, int n_in,
                              void* d_out, int out_size, void* d_ws, size_t ws_size,
                              hipStream_t stream) {
    const int*   ids   = (const int*)  d_in[0];
    const float* W_ih  = (const float*)d_in[1];
    const float* b_ih  = (const float*)d_in[2];
    const float* W_hh  = (const float*)d_in[3];
    const float* b_hh  = (const float*)d_in[4];
    const float* W_out = (const float*)d_in[5];
    const float* b_out = (const float*)d_in[6];

    char* ws = (char*)d_ws;
    u16*   whh_t  = (u16*)  (ws + OFF_WHHT);
    u16*   wout_t = (u16*)  (ws + OFF_WOUTT);
    float* bsum   = (float*)(ws + OFF_BSUM);
    u16*   hbuf   = (u16*)  (ws + OFF_HBUF);
    int*   bar    = (int*)  (ws + OFF_BAR);
    u16*   hs     = (u16*)  (ws + OFF_HS);

    // Zero h double-buffer + barrier counter every call (deterministic replay).
    hipMemsetAsync(ws + OFF_HBUF, 0, 65536 + 256, stream);

    // One-time-per-call weight conversion/transposition to bf16 [N][K].
    transpose_bf16_kernel<<<(G4H * H_ + 255) / 256, 256, 0, stream>>>(W_hh, whh_t, H_, G4H);
    transpose_bf16_kernel<<<(V_ * H_ + 255) / 256, 256, 0, stream>>>(W_out, wout_t, H_, V_);
    bsum_kernel<<<(G4H + 255) / 256, 256, 0, stream>>>(b_ih, b_hh, bsum);

    // Sequential LSTM over S steps (persistent, device-synced).
    lstm_seq_kernel<<<NBLK, 512, 0, stream>>>(ids, W_ih, whh_t, bsum, hbuf, hs, bar);

    // Output logits GEMM.
    out_gemm_kernel<<<dim3(V_ / 80, (B_ * S_) / 128), 256, 0, stream>>>(
        hs, wout_t, b_out, (float*)d_out);
}
